// SynthProxyGNN_73890617360755
// MI455X (gfx1250) — matrix operand order repaired and checked by reading, not compiled, not run on hardware
//
#include <hip/hip_runtime.h>
#include <cstdint>
#include <cstddef>

#define HID     128
#define NNODES  100000
#define NEDGES  640000
#define NGRAPHS 2048
#define NP      100096   // NNODES padded up to a multiple of 128

typedef __attribute__((ext_vector_type(16))) __bf16 bf16x16;
typedef __attribute__((ext_vector_type(8)))  float  f32x8;
typedef __attribute__((ext_vector_type(4)))  unsigned int uint4v;
typedef __attribute__((ext_vector_type(8)))  int int8v;
typedef __attribute__((ext_vector_type(4)))  int int4v;

#if __has_builtin(__builtin_amdgcn_tensor_load_to_lds) && __has_builtin(__builtin_amdgcn_s_wait_tensorcnt)
#define HAVE_TDM 1
#else
#define HAVE_TDM 0
#endif

// LDS row stride for the staged weight tile: 128 bf16 (256B) + 16B TDM pad = 136 elems
#define SWROW 136

__device__ __forceinline__ __bf16 f2bf(float f) {
  union { float f; unsigned u; } c; c.f = f;
  unsigned r = c.u + 0x7FFFu + ((c.u >> 16) & 1u);   // round-to-nearest-even
  unsigned short hs = (unsigned short)(r >> 16);
  __bf16 out; __builtin_memcpy(&out, &hs, 2); return out;
}
__device__ __forceinline__ float bf2f(__bf16 b) {
  unsigned short hs; __builtin_memcpy(&hs, &b, 2);
  union { unsigned u; float f; } c; c.u = (unsigned)hs << 16; return c.f;
}

#if HAVE_TDM
// ---------------------------------------------------------------------------
// TDM: DMA the 128x128 bf16 weight tile (row-major [N][K], contiguous) from
// global into LDS, inserting 16B of padding after every 256B row so the LDS
// row stride is 272B (bank step 4 -> conflict-free B-fragment reads).
// D# per CDNA5 ISA ch.8: group0 {count, lds_addr, global_addr, type=2},
// group1 {data_size=2B, pad_enable, pad_interval=64dw, pad_amount=4dw,
//         tensor_dim0/1=128, tile_dim0/1=128, dim0_stride=128}, groups 2/3 = 0.
// ---------------------------------------------------------------------------
__device__ __forceinline__ void tdm_load_weights(const __bf16* Wt, __bf16* sW) {
  unsigned lds_off = (unsigned)(size_t)sW;               // low 32b of flat LDS ptr = byte offset
  unsigned long long ga = (unsigned long long)(size_t)Wt;
  unsigned g0w[4];
  g0w[0] = 1u;                                           // count = 1 (valid descriptor)
  g0w[1] = lds_off;                                      // lds_addr
  g0w[2] = (unsigned)ga;                                 // global_addr[31:0]
  g0w[3] = (unsigned)((ga >> 32) & 0x01FFFFFFu) | (2u << 30);  // ga[56:32] | type=2
  unsigned g1w[8] = {0, 0, 0, 0, 0, 0, 0, 0};
  g1w[0] = (1u << 16)                                    // data_size = 2 bytes
         | (1u << 20)                                    // pad_enable
         | (5u << 22)                                    // pad_interval: 64 DWORDs (256B)
         | (3u << 25);                                   // pad_amount: 4 DWORDs (16B)
  g1w[1] = 128u << 16;                                   // tensor_dim0 = 128 (bits 79:48)
  g1w[2] = 128u << 16;                                   // tensor_dim1 = 128 (bits 111:80)
  g1w[3] = 128u << 16;                                   // tile_dim0   = 128 (bits 127:112)
  g1w[4] = 128u;                                         // tile_dim1   = 128 (bits 143:128)
  g1w[5] = 128u;                                         // tensor_dim0_stride = 128 (bits 207:160)
  uint4v g0; int8v g1;
  __builtin_memcpy(&g0, g0w, 16);
  __builtin_memcpy(&g1, g1w, 32);
  int4v gz4 = {0, 0, 0, 0};
#if defined(__clang_major__) && (__clang_major__ >= 23)
  int8v gz8 = {0, 0, 0, 0, 0, 0, 0, 0};
  __builtin_amdgcn_tensor_load_to_lds(g0, g1, gz4, gz4, gz8, 0);
#else
  __builtin_amdgcn_tensor_load_to_lds(g0, g1, gz4, gz4, 0);
#endif
  __builtin_amdgcn_s_wait_tensorcnt(0);
}
#endif

// ---------------------------------------------------------------------------
// Pack 7 weight matrices [K=128, N=128] f32 -> bf16 transposed [N][K]
// slot 0: ew2, slots 1..3: cw1[l], slots 4..6: cw2[l]
// ---------------------------------------------------------------------------
__global__ void k_pack_weights(const float* __restrict__ ew2,
                               const float* __restrict__ cw1,
                               const float* __restrict__ cw2,
                               __bf16* __restrict__ wt) {
  int t = blockIdx.x * blockDim.x + threadIdx.x;      // 7 * 16384 threads
  int m = t >> 14;
  int i = t & 16383;
  int n = i >> 7, k = i & 127;
  const float* src = (m == 0) ? ew2
                   : (m <= 3) ? cw1 + (size_t)(m - 1) * 16384
                              : cw2 + (size_t)(m - 4) * 16384;
  wt[(size_t)m * 16384 + (size_t)n * HID + k] = f2bf(src[(size_t)k * HID + n]);
}

// ---------------------------------------------------------------------------
// Node encoder: h[N,128] = x[N,7] @ node_w[7,128] + node_b
// ---------------------------------------------------------------------------
__global__ void k_node_encode(const float* __restrict__ x,
                              const float* __restrict__ w,
                              const float* __restrict__ b,
                              float* __restrict__ h) {
  int t = blockIdx.x * blockDim.x + threadIdx.x;
  if (t >= NNODES * HID) return;
  int i = t >> 7, c = t & 127;
  const float* xi = x + (size_t)i * 7;
  float acc = b[c];
#pragma unroll
  for (int k = 0; k < 7; ++k) acc += xi[k] * w[k * HID + c];
  h[t] = acc;
}

// ---------------------------------------------------------------------------
// Edge encoder stage 1: t[E,128] = relu(edge_attr[E,3] @ ew1 + eb1) in bf16
// ---------------------------------------------------------------------------
__global__ void k_edge_encode1(const float* __restrict__ ea,
                               const float* __restrict__ w,
                               const float* __restrict__ b,
                               __bf16* __restrict__ t) {
  int tid = blockIdx.x * blockDim.x + threadIdx.x;    // NEDGES*HID threads
  int i = tid >> 7, c = tid & 127;
  const float* e = ea + (size_t)i * 3;
  float acc = b[c] + e[0] * w[c] + e[1] * w[HID + c] + e[2] * w[2 * HID + c];
  t[tid] = f2bf(fmaxf(acc, 0.0f));
}

// ---------------------------------------------------------------------------
// WMMA GEMM: out[M,128] = act(A[M,128] @ W[128,128] + bias)
// W pre-packed bf16 [N][K], DMA'd into LDS once per block by the TDM (wave 0,
// TENSORcnt), then all 8 waves read B fragments from LDS (ds_load_b128).
// Block = 256 thr = 8 wave32; block tile 128x128; wave tile 16x128 = 8 accs;
// K-loop 4 x (16x16x32 bf16 WMMA).
// ---------------------------------------------------------------------------
template <typename TA, typename TO, bool RELU>
__global__ __launch_bounds__(256) void k_gemm128(const TA* __restrict__ A,
                                                 const __bf16* __restrict__ Wt,
                                                 const float* __restrict__ bias,
                                                 TO* __restrict__ out, int M) {
  __shared__ __bf16 sW[128 * SWROW];   // 34 KB: 128 rows, 272B stride

#if HAVE_TDM
  if (threadIdx.x < 32) tdm_load_weights(Wt, sW);
#else
  for (int idx = threadIdx.x; idx < 128 * HID; idx += 256) {
    int n = idx >> 7, k = idx & 127;
    sW[n * SWROW + k] = Wt[idx];
  }
#endif
  __syncthreads();

  const int lane = threadIdx.x & 31;
  const int wave = threadIdx.x >> 5;
  const int half = lane >> 4;        // 0: lanes 0-15, 1: lanes 16-31
  const int r16  = lane & 15;
  const int mrow = blockIdx.x * 128 + wave * 16;   // wave's 16-row strip
  const int arow = mrow + r16;                     // A row this lane loads

  f32x8 acc[8] = {};

#pragma unroll
  for (int kb = 0; kb < 4; ++kb) {
    const int k0 = kb * 32;
    // ----- A fragment: 16x32 bf16 (ISA layout: half0 K=[k0..k0+7],[k0+16..23];
    //                                half1 K=[k0+8..15],[k0+24..31])
    bf16x16 a;
    if constexpr (sizeof(TA) == 4) {
      float4 f0 = {0, 0, 0, 0}, f1 = f0, f2 = f0, f3 = f0;
      if (arow < M) {
        const float* ap = (const float*)A + (size_t)arow * HID + k0 + half * 8;
        f0 = *(const float4*)(ap + 0);
        f1 = *(const float4*)(ap + 4);
        f2 = *(const float4*)(ap + 16);
        f3 = *(const float4*)(ap + 20);
      }
      a[0] = f2bf(f0.x);  a[1] = f2bf(f0.y);  a[2]  = f2bf(f0.z);  a[3]  = f2bf(f0.w);
      a[4] = f2bf(f1.x);  a[5] = f2bf(f1.y);  a[6]  = f2bf(f1.z);  a[7]  = f2bf(f1.w);
      a[8] = f2bf(f2.x);  a[9] = f2bf(f2.y);  a[10] = f2bf(f2.z);  a[11] = f2bf(f2.w);
      a[12] = f2bf(f3.x); a[13] = f2bf(f3.y); a[14] = f2bf(f3.z);  a[15] = f2bf(f3.w);
    } else {
      float4 lo = {0, 0, 0, 0}, hi = lo;
      if (arow < M) {
        const __bf16* ap = (const __bf16*)A + (size_t)arow * HID + k0 + half * 8;
        lo = *(const float4*)(ap);        // 8 bf16, K = k0+8*half ..
        hi = *(const float4*)(ap + 16);   // 8 bf16, K = k0+16+8*half ..
      }
      __builtin_memcpy(&a, &lo, 16);
      __builtin_memcpy((char*)&a + 16, &hi, 16);
    }

    // ----- 8 N-tiles: B fragment 32x16 from LDS-resident Wt (two ds_load_b128)
#pragma unroll
    for (int nt = 0; nt < 8; ++nt) {
      const __bf16* bp = sW + (size_t)(nt * 16 + r16) * SWROW + k0 + half * 8;
      bf16x16 b;
      float4 b0 = *(const float4*)(bp);
      float4 b1 = *(const float4*)(bp + 16);
      __builtin_memcpy(&b, &b0, 16);
      __builtin_memcpy((char*)&b + 16, &b1, 16);
      acc[nt] = __builtin_amdgcn_wmma_f32_16x16x32_bf16(
          false, a, false, b, (short)0, acc[nt], false, false);
    }
  }

  // ----- epilogue: bias + optional ReLU; D layout: elem r -> M = 8*half + r
#pragma unroll
  for (int nt = 0; nt < 8; ++nt) {
    const int n = nt * 16 + r16;
    const float bn = bias[n];
#pragma unroll
    for (int r = 0; r < 8; ++r) {
      int m = mrow + half * 8 + r;
      if (m < M) {
        float v = acc[nt][r] + bn;
        if (RELU) v = fmaxf(v, 0.0f);
        if constexpr (sizeof(TO) == 2)
          out[(size_t)m * HID + n] = f2bf(v);
        else
          out[(size_t)m * HID + n] = v;
      }
    }
  }
}

// ---------------------------------------------------------------------------
// z = h (float4 copy)
// ---------------------------------------------------------------------------
__global__ void k_copy4(const float4* __restrict__ src, float4* __restrict__ dst, int n4) {
  int t = blockIdx.x * blockDim.x + threadIdx.x;
  if (t < n4) dst[t] = src[t];
}

// ---------------------------------------------------------------------------
// Message + scatter: z[dst] += relu(h[src] + e)   (4 channels / thread)
// ---------------------------------------------------------------------------
__global__ void k_msg_scatter(const float* __restrict__ h,
                              const __bf16* __restrict__ e,
                              const int* __restrict__ src,
                              const int* __restrict__ dst,
                              float* __restrict__ z) {
  int t = blockIdx.x * blockDim.x + threadIdx.x;    // NEDGES*32 threads
  int eid = t >> 5, c4 = (t & 31) * 4;
  int s = src[eid], d = dst[eid];
  // prefetch the streaming bf16 e operand well ahead (global_prefetch_b8)
  if (eid + 512 < NEDGES)
    __builtin_prefetch(e + (size_t)(eid + 512) * HID + c4, 0, 1);
  float4 hv = *(const float4*)(h + (size_t)s * HID + c4);
  __bf16 eb[4];
  *(float2*)eb = *(const float2*)(e + (size_t)eid * HID + c4);
  float* zp = z + (size_t)d * HID + c4;
  atomicAdd(zp + 0, fmaxf(hv.x + bf2f(eb[0]), 0.0f));
  atomicAdd(zp + 1, fmaxf(hv.y + bf2f(eb[1]), 0.0f));
  atomicAdd(zp + 2, fmaxf(hv.z + bf2f(eb[2]), 0.0f));
  atomicAdd(zp + 3, fmaxf(hv.w + bf2f(eb[3]), 0.0f));
}

// ---------------------------------------------------------------------------
// Pooling
// ---------------------------------------------------------------------------
__global__ void k_zero(float* __restrict__ p, int n) {
  int t = blockIdx.x * blockDim.x + threadIdx.x;
  if (t < n) p[t] = 0.0f;
}

__global__ void k_pool_sum(const float* __restrict__ h,
                           const int* __restrict__ batch,
                           float* __restrict__ gsum,
                           float* __restrict__ gcnt) {
  int t = blockIdx.x * blockDim.x + threadIdx.x;    // NNODES*32 threads
  int i = t >> 5, c4 = (t & 31) * 4;
  int g = batch[i];
  float4 hv = *(const float4*)(h + (size_t)i * HID + c4);
  float* gp = gsum + (size_t)g * HID + c4;
  atomicAdd(gp + 0, hv.x);
  atomicAdd(gp + 1, hv.y);
  atomicAdd(gp + 2, hv.z);
  atomicAdd(gp + 3, hv.w);
  if ((t & 31) == 0) atomicAdd(&gcnt[g], 1.0f);
}

// ---------------------------------------------------------------------------
// Heads: per-graph 128 -> 64 -> 1, twice (success / route)
// ---------------------------------------------------------------------------
__global__ __launch_bounds__(64) void k_heads(const float* __restrict__ gsum,
                                              const float* __restrict__ gcnt,
                                              const float* __restrict__ clw1,
                                              const float* __restrict__ clb1,
                                              const float* __restrict__ clw2,
                                              const float* __restrict__ clb2,
                                              const float* __restrict__ rw1,
                                              const float* __restrict__ rb1,
                                              const float* __restrict__ rw2,
                                              const float* __restrict__ rb2,
                                              float* __restrict__ out) {
  __shared__ float red[128];
  int g = blockIdx.x, j = threadIdx.x;
  float inv = 1.0f / fmaxf(gcnt[g], 1.0f);
  const float* gp = gsum + (size_t)g * HID;
  float hs = clb1[j], hr = rb1[j];
#pragma unroll 4
  for (int k = 0; k < HID; ++k) {
    float gv = gp[k] * inv;
    hs += gv * clw1[k * 64 + j];
    hr += gv * rw1[k * 64 + j];
  }
  hs = fmaxf(hs, 0.0f);
  hr = fmaxf(hr, 0.0f);
  red[j] = hs * clw2[j];
  red[64 + j] = hr * rw2[j];
  __syncthreads();
  if (j == 0) {
    float s = clb2[0];
    for (int k = 0; k < 64; ++k) s += red[k];
    out[g] = s;
  } else if (j == 1) {
    float s = rb2[0];
    for (int k = 0; k < 64; ++k) s += red[64 + k];
    out[NGRAPHS + g] = s;
  }
}

// ---------------------------------------------------------------------------
extern "C" void kernel_launch(void* const* d_in, const int* in_sizes, int n_in,
                              void* d_out, int out_size, void* d_ws, size_t ws_size,
                              hipStream_t stream) {
  (void)in_sizes; (void)n_in; (void)out_size; (void)ws_size;

  const float* x         = (const float*)d_in[0];
  const float* edge_attr = (const float*)d_in[1];
  const int*   eidx      = (const int*)d_in[2];
  const int*   batch     = (const int*)d_in[3];
  const float* node_w    = (const float*)d_in[4];
  const float* node_b    = (const float*)d_in[5];
  const float* ew1       = (const float*)d_in[6];
  const float* eb1       = (const float*)d_in[7];
  const float* ew2       = (const float*)d_in[8];
  const float* eb2       = (const float*)d_in[9];
  const float* cw1       = (const float*)d_in[10];
  const float* cb1       = (const float*)d_in[11];
  const float* cw2       = (const float*)d_in[12];
  const float* cb2       = (const float*)d_in[13];
  const float* clw1      = (const float*)d_in[14];
  const float* clb1      = (const float*)d_in[15];
  const float* clw2      = (const float*)d_in[16];
  const float* clb2      = (const float*)d_in[17];
  const float* rw1       = (const float*)d_in[18];
  const float* rb1       = (const float*)d_in[19];
  const float* rw2       = (const float*)d_in[20];
  const float* rb2       = (const float*)d_in[21];

  const int* esrc = eidx;
  const int* edst = eidx + NEDGES;

  // ---- workspace carve-up (256B aligned) ----
  char* ws = (char*)d_ws;
  auto up = [](size_t v) { return (v + 255) & ~(size_t)255; };
  size_t off = 0;
  __bf16* wt   = (__bf16*)(ws + off); off += up((size_t)7 * 16384 * 2);          // packed weights
  float*  hbuf = (float*)(ws + off);  off += up((size_t)NP * HID * 4);           // h
  float*  zbuf = (float*)(ws + off);  off += up((size_t)NP * HID * 4);           // z = h + agg
  float*  g1   = (float*)(ws + off);  off += up((size_t)NP * HID * 4);           // MLP mid
  __bf16* tbuf = (__bf16*)(ws + off); off += up((size_t)NEDGES * HID * 2);       // edge mid (bf16)
  __bf16* ebuf = (__bf16*)(ws + off); off += up((size_t)NEDGES * HID * 2);       // e (bf16)
  float*  gsum = (float*)(ws + off);  off += (size_t)NGRAPHS * HID * 4;          // pool sum
  float*  gcnt = (float*)(ws + off);  off += (size_t)NGRAPHS * 4;                // pool count

  // ---- encoders + weight packing ----
  k_pack_weights<<<(7 * 16384) / 256, 256, 0, stream>>>(ew2, cw1, cw2, wt);
  k_node_encode<<<(NNODES * HID + 255) / 256, 256, 0, stream>>>(x, node_w, node_b, hbuf);
  k_edge_encode1<<<(NEDGES * HID) / 256, 256, 0, stream>>>(edge_attr, ew1, eb1, tbuf);

  // e = t @ ew2 + eb2   (bf16 in, bf16 out, no relu)  -- WMMA + TDM
  k_gemm128<__bf16, __bf16, false><<<NEDGES / 128, 256, 0, stream>>>(tbuf, wt, eb2, ebuf, NEDGES);

  const int nodeGrid = (NNODES + 127) / 128;   // 782
  for (int l = 0; l < 3; ++l) {
    k_copy4<<<(NNODES * HID / 4) / 256, 256, 0, stream>>>((const float4*)hbuf, (float4*)zbuf,
                                                          NNODES * HID / 4);
    k_msg_scatter<<<(NEDGES * 32) / 256, 256, 0, stream>>>(hbuf, ebuf, esrc, edst, zbuf);
    // g1 = relu(z @ cw1[l] + cb1[l]); h = relu(g1 @ cw2[l] + cb2[l])  -- WMMA + TDM
    k_gemm128<float, float, true><<<nodeGrid, 256, 0, stream>>>(
        zbuf, wt + (size_t)(1 + l) * 16384, cb1 + l * HID, g1, NNODES);
    k_gemm128<float, float, true><<<nodeGrid, 256, 0, stream>>>(
        g1, wt + (size_t)(4 + l) * 16384, cb2 + l * HID, hbuf, NNODES);
  }

  // ---- pool + heads ----
  k_zero<<<(NGRAPHS * HID + NGRAPHS) / 256, 256, 0, stream>>>(gsum, NGRAPHS * HID + NGRAPHS);
  k_pool_sum<<<(NNODES * 32) / 256, 256, 0, stream>>>(hbuf, batch, gsum, gcnt);
  k_heads<<<NGRAPHS, 64, 0, stream>>>(gsum, gcnt, clw1, clb1, clw2, clb2,
                                      rw1, rb1, rw2, rb2, (float*)d_out);
}
